// PairBackflow_1374389535087
// MI455X (gfx1250) — matrix operand order repaired and checked by reading, not compiled, not run on hardware
//
#include <hip/hip_runtime.h>

// CDNA5 / gfx1250, wave32. One v_wmma_f32_16x16x32_f16 per 16-pair tile
// implements the 16x16 hidden layer of the per-pair eta MLP.
// tanh via hardware V_TANH_F32; cross-lane reduction via DPP8/permlane16
// (VALU pipe) + one ds_bpermute gather. Simple (non-pipelined) loop keeps
// VGPRs ~125 -> 8 waves/SIMD, letting wave-level parallelism hide the
// WMMA hazard window and the bpermute latency (pipelining caused spills).

typedef __attribute__((ext_vector_type(16))) _Float16 v16h;
typedef __attribute__((ext_vector_type(8)))  float    v8f;

#define HID            16
#define TILE_J         16
#define WAVES_PER_BLK  8
#define BLK_THREADS    256
#define MAX_SLICES     16

// dpp8 selector: 3 bits per lane, lane0 in bits [2:0]
#define DPP8SEL(a0,a1,a2,a3,a4,a5,a6,a7) \
  ((a0)|((a1)<<3)|((a2)<<6)|((a3)<<9)|((a4)<<12)|((a5)<<15)|((a6)<<18)|((a7)<<21))

__device__ __forceinline__ float xor1f(float v) {
#if __has_builtin(__builtin_amdgcn_mov_dpp8)
  return __int_as_float(__builtin_amdgcn_mov_dpp8(
      __float_as_int(v), DPP8SEL(1,0,3,2,5,4,7,6)));
#else
  return __shfl_xor(v, 1, 32);
#endif
}
__device__ __forceinline__ float xor2f(float v) {
#if __has_builtin(__builtin_amdgcn_mov_dpp8)
  return __int_as_float(__builtin_amdgcn_mov_dpp8(
      __float_as_int(v), DPP8SEL(2,3,0,1,6,7,4,5)));
#else
  return __shfl_xor(v, 2, 32);
#endif
}
__device__ __forceinline__ float xor4f(float v) {
#if __has_builtin(__builtin_amdgcn_mov_dpp8)
  return __int_as_float(__builtin_amdgcn_mov_dpp8(
      __float_as_int(v), DPP8SEL(4,5,6,7,0,1,2,3)));
#else
  return __shfl_xor(v, 4, 32);
#endif
}
__device__ __forceinline__ float xor8f(float v) {
#if __has_builtin(__builtin_amdgcn_permlane16)
  int i = __float_as_int(v);
  // per-lane 4-bit selectors: lane l reads lane l^8 within its 16-lane row
  return __int_as_float(__builtin_amdgcn_permlane16(
      i, i, 0xFEDCBA98u, 0x76543210u, false, false));
#else
  return __shfl_xor(v, 8, 32);
#endif
}

// Branch-free tanh. gfx1250 has V_TANH_F32; otherwise use
// tanh(|x|) = 1 - 2/(exp2(2|x|*log2e)+1), sign restored with v_bfi.
__device__ __forceinline__ float fast_tanh(float v) {
#if __has_builtin(__builtin_amdgcn_tanhf)
  return __builtin_amdgcn_tanhf(v);
#else
  const float LOG2E_X2 = 2.8853900817779268f;       // 2*log2(e)
  float e = __builtin_amdgcn_exp2f(LOG2E_X2 * fabsf(v));
  float r = __builtin_amdgcn_rcpf(e + 1.0f);        // e->inf => r->0 => t->1
  float t = fmaf(-2.0f, r, 1.0f);
  return copysignf(t, v);
#endif
}

__global__ __launch_bounds__(BLK_THREADS)
void backflow_pair_kernel(const float* __restrict__ x,
                          const float* __restrict__ cell,
                          const float* __restrict__ Wp1, const float* __restrict__ bp1,
                          const float* __restrict__ Wp2, const float* __restrict__ bp2,
                          const float* __restrict__ Wp3, const float* __restrict__ bp3,
                          const float* __restrict__ Wa1, const float* __restrict__ ba1,
                          const float* __restrict__ Wa2, const float* __restrict__ ba2,
                          const float* __restrict__ Wa3, const float* __restrict__ ba3,
                          const int*   __restrict__ n_up_p,
                          float* __restrict__ partial,
                          int n, int slice_len)
{
  const int jt   = blockIdx.x;           // j-tile index (16 rows)
  const int sl   = blockIdx.y;           // i-slice index
  const int j0   = jt * TILE_J;
  const int i0   = sl * slice_len;
  const int tid  = threadIdx.x;
  const int wave = tid >> 5;
  const int lane = tid & 31;
  const int m    = lane & 15;            // row within j-tile; also N index of WMMA C
  const bool hiHalf = lane >= 16;        // lanes 16..31 hold K=8..15 of the A fragment

  __shared__ float part[WAVES_PER_BLK][TILE_J][3];

  // Warm weight cachelines (gfx1250 global_prefetch path).
  __builtin_prefetch(Wp2, 0, 0);
  __builtin_prefetch(Wa2, 0, 0);

  // ---- uniform setup: 3x3 cell inverse, r_s of the HEG ----
  const float c00 = cell[0], c01 = cell[1], c02 = cell[2];
  const float c10 = cell[3], c11 = cell[4], c12 = cell[5];
  const float c20 = cell[6], c21 = cell[7], c22 = cell[8];
  const float det = c00*(c11*c22 - c12*c21)
                  - c01*(c10*c22 - c12*c20)
                  + c02*(c10*c21 - c11*c20);
  const float invdet = 1.0f / det;
  const float i00 =  (c11*c22 - c12*c21) * invdet;
  const float i01 = -(c01*c22 - c02*c21) * invdet;
  const float i02 =  (c01*c12 - c02*c11) * invdet;
  const float i10 = -(c10*c22 - c12*c20) * invdet;
  const float i11 =  (c00*c22 - c02*c20) * invdet;
  const float i12 = -(c00*c12 - c02*c10) * invdet;
  const float i20 =  (c10*c21 - c11*c20) * invdet;
  const float i21 = -(c00*c21 - c01*c20) * invdet;
  const float i22 =  (c00*c11 - c01*c10) * invdet;

  const float PI    = 3.14159265358979323846f;
  const float invPI = 0.31830988618379067154f;
  const float vol_pp = fabsf(det) / (float)n;
  // rs = (4*pi/(3*vol_pp))^(-1/3)  =>  1/rs = cbrt(4*pi/(3*vol_pp))
  const float inv_rs = cbrtf((4.0f * PI) / (3.0f * vol_pp));

  const int nup = *n_up_p;
  const bool jspin_up = (j0 < nup);      // 16 | n_up, so a j-tile never straddles spins

  // booleans / gather lane for the transpose-reduce butterfly
  const bool b0  = (lane & 1) != 0;
  const bool bb1 = (lane & 2) != 0;
  const bool bb2 = (lane & 4) != 0;
  // after the butterfly, row r (0..7) lives in lanes 0..15 at lane&7==r,
  // row r+8 in lanes 16..31 at lane&7==r; lane l wants row l&15:
  const int gatherLane = (m < 8) ? m : (m + 8);

  // ---- per-lane register-resident parameters for BOTH spin channels ----
  // Layer-1: half-wave split of the 16 hidden units matches the f16 A layout.
  const int koff = hiHalf ? 8 : 0;
  float w1P[8], b1P[8], w1A[8], b1A[8];
#pragma unroll
  for (int e = 0; e < 8; ++e) {
    w1P[e] = Wp1[koff + e];  b1P[e] = bp1[koff + e];
    w1A[e] = Wa1[koff + e];  b1A[e] = ba1[koff + e];
  }
  // Layer-2 B fragment: B[K=k][N=n] = W2[k][n]; elements 0..7 hold K=koff+e
  // (matching A), elements 8..15 hold the K=16..31 zero padding.
  v16h bP, bA;
#pragma unroll
  for (int e = 0; e < 16; ++e) {
    bP[e] = (e >= 8) ? (_Float16)0.0f : (_Float16)Wp2[(koff + e) * HID + m];
    bA[e] = (e >= 8) ? (_Float16)0.0f : (_Float16)Wa2[(koff + e) * HID + m];
  }
  // Layer-2 bias folded into the WMMA C operand (C[m][n] = b2[n], n = lane&15).
  const float b2vP = bp2[m], b2vA = ba2[m];
  v8f cbP, cbA;
#pragma unroll
  for (int r = 0; r < 8; ++r) { cbP[r] = b2vP; cbA[r] = b2vA; }
  const float w3P = Wp3[m], w3A = Wa3[m];
  const float b3P = bp3[0], b3A = ba3[0];

  // ---- per-lane j-row features (rows j0..j0+15; both halves mirror) ----
  const int j  = j0 + m;
  const float pjx = x[j * 3 + 0], pjy = x[j * 3 + 1], pjz = x[j * 3 + 2];
  const float fjx = pjx * i00 + pjy * i10 + pjz * i20;   // frac = x @ inv(cell)
  const float fjy = pjx * i01 + pjy * i11 + pjz * i21;
  const float fjz = pjx * i02 + pjy * i12 + pjz * i22;

  float accx = 0.0f, accy = 0.0f, accz = 0.0f;

  // loop-invariant bound (hoisted); wave-uniform trip counts
  const int rem   = n - i0;
  const int iiend = (slice_len < rem) ? slice_len : rem;

  for (int ii = wave; ii < iiend; ii += WAVES_PER_BLK) {
    const int i = i0 + ii;

    // spin-channel select (uniform across the wave each iteration)
    const bool smp = (jspin_up == (i < nup));

    // frac_i (uniform address -> broadcast loads)
    const float pix = x[i * 3 + 0], piy = x[i * 3 + 1], piz = x[i * 3 + 2];
    const float fix = pix * i00 + piy * i10 + piz * i20;
    const float fiy = pix * i01 + piy * i11 + piz * i21;
    const float fiz = pix * i02 + piy * i12 + piz * i22;

    // half-sinusoid periodic displacement: sin(pi*dfrac) @ cell / pi
    const float s0 = __sinf(PI * (fjx - fix));
    const float s1 = __sinf(PI * (fjy - fiy));
    const float s2 = __sinf(PI * (fjz - fiz));
    const float hx = (s0 * c00 + s1 * c10 + s2 * c20) * invPI;
    const float hy = (s0 * c01 + s1 * c11 + s2 * c21) * invPI;
    const float hz = (s0 * c02 + s1 * c12 + s2 * c22) * invPI;

    const bool diag = (i == j);
    const float dist2 = hx * hx + hy * hy + hz * hz + (diag ? 1.0f : 0.0f);
    const float s = sqrtf(dist2) * inv_rs;                    // dist / r_s

    // Layer 1 (per pair, VALU+TRANS): A[m][k] = tanh(s*W1[k]+b1[k]); K 16..31 = 0.
    v16h a;
#pragma unroll
    for (int e = 0; e < 8; ++e) {
      const float w1e = smp ? w1P[e] : w1A[e];
      const float b1e = smp ? b1P[e] : b1A[e];
      a[e] = (_Float16)fast_tanh(fmaf(s, w1e, b1e));
    }
#pragma unroll
    for (int e = 8; e < 16; ++e) a[e] = (_Float16)0.0f;

    // Layer 2 (tensor core): C = A(16x32,f16) x W2(32x16,f16) + b2
    const v16h bfrag = smp ? bP : bA;
    const v8f  cbias = smp ? cbP : cbA;
    v8f c = __builtin_amdgcn_wmma_f32_16x16x32_f16(
        false, a, false, bfrag, (short)0, cbias, false, false);

    // Layer 3: eta[m] = sum_n tanh(C[m][n]) * W3[n] + b3.
    // Transpose-reduce butterfly over the 16-lane N group (DPP/permlane, no LDS):
    const float w3v = smp ? w3P : w3A;
    const float b3v = smp ? b3P : b3A;
    float t0 = fast_tanh(c[0]) * w3v, t1 = fast_tanh(c[1]) * w3v;
    float t2 = fast_tanh(c[2]) * w3v, t3 = fast_tanh(c[3]) * w3v;
    float t4 = fast_tanh(c[4]) * w3v, t5 = fast_tanh(c[5]) * w3v;
    float t6 = fast_tanh(c[6]) * w3v, t7 = fast_tanh(c[7]) * w3v;
    // stage 1 (xor 1): 8 regs -> 4; row = 2p + bit0(lane)
    float u0 = (b0 ? t1 : t0) + xor1f(b0 ? t0 : t1);
    float u1 = (b0 ? t3 : t2) + xor1f(b0 ? t2 : t3);
    float u2 = (b0 ? t5 : t4) + xor1f(b0 ? t4 : t5);
    float u3 = (b0 ? t7 : t6) + xor1f(b0 ? t6 : t7);
    // stage 2 (xor 2): 4 -> 2; row = 4q + 2*bit1 + bit0
    float v0 = (bb1 ? u1 : u0) + xor2f(bb1 ? u0 : u1);
    float v1 = (bb1 ? u3 : u2) + xor2f(bb1 ? u2 : u3);
    // stage 3 (xor 4): 2 -> 1; row = lane&7 (+8 in the hi half)
    float z  = (bb2 ? v1 : v0) + xor4f(bb2 ? v0 : v1);
    // stage 4 (xor 8): complete the 16-column sum
    z += xor8f(z);
    // gather row m into lane m (single bpermute) and finish
    float eta_m = __shfl(z, gatherLane, 32) + b3v;
    if (diag) eta_m = 0.0f;          // (1 - eye) mask

    // dr[j] += eta[j,i] * d_hsin[j,i]   (eta symmetric, d_hsin antisymmetric)
    accx = fmaf(eta_m, hx, accx);
    accy = fmaf(eta_m, hy, accy);
    accz = fmaf(eta_m, hz, accz);
  }

  // ---- cross-wave reduction of the 16-row partial dr through LDS ----
  if (lane < 16) {
    part[wave][m][0] = accx;
    part[wave][m][1] = accy;
    part[wave][m][2] = accz;
  }
  __syncthreads();

  if (tid < TILE_J * 3) {
    const int mm = tid / 3, cc = tid % 3;
    float ssum = 0.0f;
#pragma unroll
    for (int w = 0; w < WAVES_PER_BLK; ++w) ssum += part[w][mm][cc];
    partial[((size_t)sl * n + (j0 + mm)) * 3 + cc] = ssum;
  }
}

__global__ void backflow_reduce_kernel(const float* __restrict__ x,
                                       const float* __restrict__ partial,
                                       float* __restrict__ out,
                                       int n, int nslices)
{
  const int e = blockIdx.x * blockDim.x + threadIdx.x;
  if (e < n * 3) {
    float s = x[e];
    for (int sl = 0; sl < nslices; ++sl) s += partial[(size_t)sl * n * 3 + e];
    out[e] = s;
  }
}

extern "C" void kernel_launch(void* const* d_in, const int* in_sizes, int n_in,
                              void* d_out, int out_size, void* d_ws, size_t ws_size,
                              hipStream_t stream) {
  const float* x    = (const float*)d_in[0];
  const float* cell = (const float*)d_in[1];
  const float* Wp1  = (const float*)d_in[2];
  const float* bp1  = (const float*)d_in[3];
  const float* Wp2  = (const float*)d_in[4];
  const float* bp2  = (const float*)d_in[5];
  const float* Wp3  = (const float*)d_in[6];
  const float* bp3  = (const float*)d_in[7];
  const float* Wa1  = (const float*)d_in[8];
  const float* ba1  = (const float*)d_in[9];
  const float* Wa2  = (const float*)d_in[10];
  const float* ba2  = (const float*)d_in[11];
  const float* Wa3  = (const float*)d_in[12];
  const float* ba3  = (const float*)d_in[13];
  const int*   nup  = (const int*)d_in[14];

  const int n = in_sizes[0] / 3;           // 1536

  // i-slices sized to fit partial dr buffers in the workspace.
  const size_t per_slice = (size_t)n * 3 * sizeof(float);
  int S = MAX_SLICES;
  if (ws_size < (size_t)S * per_slice) {
    S = (int)(ws_size / per_slice);
    if (S < 1) S = 1;
  }
  const int slice_len = (n + S - 1) / S;
  float* partial = (float*)d_ws;

  dim3 grid(n / TILE_J, S);
  backflow_pair_kernel<<<grid, BLK_THREADS, 0, stream>>>(
      x, cell, Wp1, bp1, Wp2, bp2, Wp3, bp3,
      Wa1, ba1, Wa2, ba2, Wa3, ba3, nup,
      partial, n, slice_len);

  const int tot = n * 3;
  backflow_reduce_kernel<<<(tot + 255) / 256, 256, 0, stream>>>(
      x, partial, (float*)d_out, n, S);
}
